// LearnedFlowUpsample3D_90297392431337
// MI455X (gfx1250) — compile-verified, matrix operand order checked by reading, not run on hardware
//
#include <hip/hip_runtime.h>
#include <hip/hip_bf16.h>

typedef __attribute__((ext_vector_type(16))) _Float16 v16h;
typedef __attribute__((ext_vector_type(8)))  _Float16 v8h;
typedef __attribute__((ext_vector_type(8)))  float    v8f;

#define BDIM 2
#define CDIM 3
#define DHW 110592           // 48*48*48
#define HID 64
#define OCH 216              // 8*27
#define OCH_PAD 224          // 14 tiles of 16
#define NTILES 14
#define EPS 1e-5f

union U16H { v16h v; _Float16 e[16]; };
union U8H  { v8h  v; _Float16 e[8]; };
union U8F  { v8f  v; float    e[8]; };

// ---------------------------------------------------------------------------
// Kernel 1: 3x3x3 conv (3->64, zero 'SAME' pad) + per-(b,c) sum / sumsq stats.
// h stored channel-minor: h_t[b][pos][64]  (contiguous 64-ch rows for WMMA B).
// grid: (432 spatial blocks, 64 out-ch, 2 batch), block 256.
// ---------------------------------------------------------------------------
__global__ void __launch_bounds__(256)
conv_stats_kernel(const float* __restrict__ flow, const float* __restrict__ W1,
                  float* __restrict__ h_t, float* __restrict__ stats) {
  const int tid = threadIdx.x;
  const int pos = blockIdx.x * 256 + tid;
  const int co  = blockIdx.y;
  const int b   = blockIdx.z;

  __shared__ float sW[81];
  __shared__ float red1[256];
  __shared__ float red2[256];
  if (tid < 81) sW[tid] = W1[co * 81 + tid];
  __syncthreads();

  const int z = pos / 2304;
  const int r = pos % 2304;
  const int y = r / 48;
  const int x = r % 48;

  const float* fb = flow + (size_t)b * CDIM * DHW;
  float acc = 0.f;
  #pragma unroll
  for (int kz = 0; kz < 3; ++kz) {
    const int zz = z + kz - 1;
    if ((unsigned)zz >= 48u) continue;
    #pragma unroll
    for (int ky = 0; ky < 3; ++ky) {
      const int yy = y + ky - 1;
      if ((unsigned)yy >= 48u) continue;
      #pragma unroll
      for (int kx = 0; kx < 3; ++kx) {
        const int xx = x + kx - 1;
        if ((unsigned)xx >= 48u) continue;
        const int sp = zz * 2304 + yy * 48 + xx;
        const int kk = kz * 9 + ky * 3 + kx;
        acc += fb[sp]            * sW[kk]
             + fb[DHW + sp]      * sW[27 + kk]
             + fb[2 * DHW + sp]  * sW[54 + kk];
      }
    }
  }
  h_t[((size_t)b * DHW + pos) * HID + co] = acc;

  red1[tid] = acc;
  red2[tid] = acc * acc;
  __syncthreads();
  #pragma unroll
  for (int s = 128; s > 0; s >>= 1) {
    if (tid < s) { red1[tid] += red1[tid + s]; red2[tid] += red2[tid + s]; }
    __syncthreads();
  }
  if (tid == 0) {
    atomicAdd(&stats[(b * HID + co) * 2 + 0], red1[0]);
    atomicAdd(&stats[(b * HID + co) * 2 + 1], red2[0]);
  }
}

// ---------------------------------------------------------------------------
// Kernel 2: fold InstanceNorm (+affine) into per-(b,c) scale/shift pairs.
// ---------------------------------------------------------------------------
__global__ void stats_finalize(const float* __restrict__ stats,
                               const float* __restrict__ gamma,
                               const float* __restrict__ beta,
                               float* __restrict__ sshift) {
  const int i = threadIdx.x;  // 0..127 = b*64+c
  if (i < BDIM * HID) {
    const int c = i & (HID - 1);
    const float inv_n = 1.f / (float)DHW;
    const float mean = stats[i * 2 + 0] * inv_n;
    const float var  = stats[i * 2 + 1] * inv_n - mean * mean;
    const float rstd = rsqrtf(var + EPS);
    const float sc   = gamma[c] * rstd;
    sshift[i * 2 + 0] = sc;
    sshift[i * 2 + 1] = beta[c] - mean * sc;
  }
}

// ---------------------------------------------------------------------------
// Kernel 3: W2 f32[216][64] -> f16[224][64] (zero-padded rows).
// ---------------------------------------------------------------------------
__global__ void w2_convert(const float* __restrict__ W2, _Float16* __restrict__ w2h) {
  const int idx = blockIdx.x * 256 + threadIdx.x;
  if (idx < OCH_PAD * HID) {
    const int o = idx >> 6;
    w2h[idx] = (o < OCH) ? (_Float16)W2[idx] : (_Float16)0.f;
  }
}

// ---------------------------------------------------------------------------
// Kernel 4: fused  (norm+LeakyReLU) -> WMMA 64->216 GEMM -> bias -> softmax(27)
//           -> convex neighbor gather (edge-clamp, x2) -> pixel shuffle.
// 1 wave = 16 spatial positions; 4 waves per block; grid (6912/4, B).
// ---------------------------------------------------------------------------
#define WAVES 4
#define SMROW 17              // padded LDS row stride
__global__ void __launch_bounds__(32 * WAVES)
fused_kernel(const float* __restrict__ flow, const float* __restrict__ h_t,
             const float* __restrict__ sshift, const _Float16* __restrict__ w2h,
             const float* __restrict__ b2, float* __restrict__ out) {
  __shared__ float smask[WAVES][OCH_PAD * SMROW];
  __shared__ float2 ssh[HID];               // {scale, shift} per channel, this batch
  __shared__ float  sb2[OCH];

  const int lane = threadIdx.x & 31;
  const int wv   = threadIdx.x >> 5;
  const int b    = blockIdx.y;
  const int tile = blockIdx.x * WAVES + wv;   // 16-position tile id, < 6912
  const int posBase = tile * 16;
  const int n    = lane & 15;                 // spatial column within tile / A-row
  const int half = lane >> 4;

  if (threadIdx.x < HID) {
    const float2 p = ((const float2*)sshift)[b * HID + threadIdx.x];
    ssh[threadIdx.x] = p;
  }
  for (int i = threadIdx.x; i < OCH; i += 32 * WAVES) sb2[i] = b2[i];
  __syncthreads();

  // ---- build B operands directly at the lane's K-chunks (no register-array
  //      dynamic indexing: memory/LDS handle the runtime `half` offset) ----
  // 16-bit 16x32 operand layout: lane-half h owns K = c*16 + half*8 + s,
  // slot = (c&1)*8 + s of operand c>>1.
  const float* hbase = h_t + ((size_t)b * DHW + posBase + n) * HID;
  U16H ub0, ub1;
  #pragma unroll
  for (int c = 0; c < 4; ++c) {
    const int kb = c * 16 + half * 8;          // runtime (half), used as address
    const float4 f0 = *(const float4*)(hbase + kb);
    const float4 f1 = *(const float4*)(hbase + kb + 4);
    const float vals[8] = { f0.x, f0.y, f0.z, f0.w, f1.x, f1.y, f1.z, f1.w };
    #pragma unroll
    for (int s = 0; s < 8; ++s) {
      const float2 ss = ssh[kb + s];           // dynamic LDS index: native
      float v = vals[s] * ss.x + ss.y;
      v = v >= 0.f ? v : 0.1f * v;
      const _Float16 hv = (_Float16)v;
      if      (c == 0) ub0.e[s]     = hv;
      else if (c == 1) ub0.e[8 + s] = hv;
      else if (c == 2) ub1.e[s]     = hv;
      else             ub1.e[8 + s] = hv;
    }
  }

  float* sm = &smask[wv][0];

  // ---- 14 output-channel tiles: D(16x16) = W2tile(16x32) x h(32x16), K=64 ----
  for (int t = 0; t < NTILES; ++t) {
    const _Float16* arow = w2h + (t * 16 + n) * HID;   // A row m = lane&15
    U8H c0, c1, c2, c3;
    c0.v = *(const v8h*)(arow +      half * 8);
    c1.v = *(const v8h*)(arow + 16 + half * 8);
    c2.v = *(const v8h*)(arow + 32 + half * 8);
    c3.v = *(const v8h*)(arow + 48 + half * 8);
    U16H ua0, ua1;
    #pragma unroll
    for (int s = 0; s < 8; ++s) {
      ua0.e[s] = c0.e[s]; ua0.e[8 + s] = c1.e[s];
      ua1.e[s] = c2.e[s]; ua1.e[8 + s] = c3.e[s];
    }
    v8f acc = {};
    acc = __builtin_amdgcn_wmma_f32_16x16x32_f16(false, ua0.v, false, ub0.v,
                                                 (short)0, acc, false, false);
    acc = __builtin_amdgcn_wmma_f32_16x16x32_f16(false, ua1.v, false, ub1.v,
                                                 (short)0, acc, false, false);
    U8F uc; uc.v = acc;
    // C layout: lane half picks M = v + 8*half, N = lane&15
    #pragma unroll
    for (int v = 0; v < 8; ++v) {
      const int ro = t * 16 + v + 8 * half;
      if (ro < OCH) sm[ro * SMROW + n] = uc.e[v] + sb2[ro];
    }
  }
  __syncthreads();

  // ---- softmax over 27 + convex neighbor gather + pixel shuffle ----
  const float* fb = flow + (size_t)b * CDIM * DHW;
  #pragma unroll
  for (int it = 0; it < 4; ++it) {
    const int tsk = it * 32 + lane;       // 0..127 = 16 positions x 8 sub-voxels
    const int i   = tsk >> 4;             // sub-voxel index (sz,sy,sx)
    const int nn  = tsk & 15;
    const int posn = posBase + nn;
    const int z = posn / 2304;
    const int r = posn % 2304;
    const int y = r / 48;
    const int x = r % 48;

    float w[27];
    int   off[27];
    float mx = -1e30f;
    int k = 0;
    #pragma unroll
    for (int dz = 0; dz < 3; ++dz) {
      const int zz = min(max(z + dz - 1, 0), 47);
      #pragma unroll
      for (int dy = 0; dy < 3; ++dy) {
        const int yy = min(max(y + dy - 1, 0), 47);
        #pragma unroll
        for (int dx = 0; dx < 3; ++dx) {
          const int xx = min(max(x + dx - 1, 0), 47);
          off[k] = zz * 2304 + yy * 48 + xx;
          const float m = sm[(i * 27 + k) * SMROW + nn];
          mx = fmaxf(mx, m);
          w[k] = m;
          ++k;
        }
      }
    }
    float s = 0.f;
    #pragma unroll
    for (k = 0; k < 27; ++k) { w[k] = __expf(w[k] - mx); s += w[k]; }
    const float inv = 2.0f / s;           // folds the *S neighbor scale

    const int sz = i >> 2, sy = (i >> 1) & 1, sx = i & 1;
    #pragma unroll
    for (int c = 0; c < CDIM; ++c) {
      const float* fc = fb + c * DHW;
      float acc = 0.f;
      #pragma unroll
      for (k = 0; k < 27; ++k) acc += w[k] * fc[off[k]];
      out[((((size_t)b * CDIM + c) * 96 + 2 * z + sz) * 96 + 2 * y + sy) * 96
          + 2 * x + sx] = acc * inv;
    }
  }
}

// ---------------------------------------------------------------------------
extern "C" void kernel_launch(void* const* d_in, const int* in_sizes, int n_in,
                              void* d_out, int out_size, void* d_ws, size_t ws_size,
                              hipStream_t stream) {
  const float* flow  = (const float*)d_in[0];
  const float* W1    = (const float*)d_in[1];
  const float* gamma = (const float*)d_in[2];
  const float* beta  = (const float*)d_in[3];
  const float* W2    = (const float*)d_in[4];
  const float* b2    = (const float*)d_in[5];
  float* out = (float*)d_out;

  char* ws = (char*)d_ws;
  float*    stats  = (float*)(ws);            // 256 f32
  float*    sshift = (float*)(ws + 1024);     // 256 f32 (scale,shift per b,c)
  _Float16* w2h    = (_Float16*)(ws + 2048);  // 224*64 f16 = 28672 B
  float*    h_t    = (float*)(ws + 32768);    // B*110592*64 f32 = 56.6 MB

  hipMemsetAsync(stats, 0, 1024, stream);

  dim3 g1(DHW / 256, HID, BDIM);
  conv_stats_kernel<<<g1, 256, 0, stream>>>(flow, W1, h_t, stats);

  stats_finalize<<<1, 128, 0, stream>>>(stats, gamma, beta, sshift);

  w2_convert<<<(OCH_PAD * HID + 255) / 256, 256, 0, stream>>>(W2, w2h);

  dim3 g4((DHW / 16) / WAVES, BDIM);
  fused_kernel<<<g4, 32 * WAVES, 0, stream>>>(flow, h_t, sshift, w2h, b2, out);
}
